// FESTA_4398046511719
// MI455X (gfx1250) — matrix operand order repaired and checked by reading, not compiled, not run on hardware
//
#include <hip/hip_runtime.h>
#include <cstdint>
#include <cstddef>

typedef _Float16 half_t;
typedef __attribute__((ext_vector_type(16))) _Float16 v16h;
typedef __attribute__((ext_vector_type(8)))  float   v8f;
typedef __attribute__((ext_vector_type(4)))  float   v4f;

#define Bc 8
#define Nc 4096
#define TPB 256

static inline int cdivh(long long a, long long b){ return (int)((a + b - 1) / b); }
static inline int ruh(int x, int m){ return ((x + m - 1) / m) * m; }

// ======================= kernels =======================

// [B,3,n] -> [B,n,3]
__global__ void k_tin(const float* __restrict__ s, float* __restrict__ d, int n){
  int t = blockIdx.x * blockDim.x + threadIdx.x;
  if (t >= Bc * 3 * n) return;
  int i = t % n, c = (t / n) % 3, b = t / (3 * n);
  d[((size_t)b * n + i) * 3 + c] = s[(size_t)b * 3 * n + (size_t)c * n + i];
}

// [B,n,3] -> [B,3,n]
__global__ void k_tout3(const float* __restrict__ s, float* __restrict__ d, int n){
  int t = blockIdx.x * blockDim.x + threadIdx.x;
  if (t >= Bc * 3 * n) return;
  int i = t % n, c = (t / n) % 3, b = t / (3 * n);
  d[(size_t)b * 3 * n + (size_t)c * n + i] = s[((size_t)b * n + i) * 3 + c];
}

// farthest point sampling: one block per batch, dist kept in LDS
__global__ void k_fps(const float* __restrict__ xyz, int n, int np, int* __restrict__ out){
  __shared__ float dist[4096];
  __shared__ float rv[TPB];
  __shared__ int   ri[TPB];
  int b = blockIdx.x, tid = threadIdx.x;
  const float* X = xyz + (size_t)b * n * 3;
  for (int j = tid; j < n; j += TPB) dist[j] = 1e10f;
  __syncthreads();
  int far = 0;
  for (int i = 0; i < np; i++){
    if (tid == 0) out[b * np + i] = far;
    float cx = X[far * 3 + 0], cy = X[far * 3 + 1], cz = X[far * 3 + 2];
    float best = -1.0f; int bj = 0;
    for (int j = tid; j < n; j += TPB){
      float dx = X[j * 3 + 0] - cx, dy = X[j * 3 + 1] - cy, dz = X[j * 3 + 2] - cz;
      float dd = dx * dx + dy * dy + dz * dz;
      float nd = fminf(dist[j], dd);
      dist[j] = nd;
      if (nd > best){ best = nd; bj = j; }
    }
    rv[tid] = best; ri[tid] = bj;
    __syncthreads();
    for (int off = TPB / 2; off > 0; off >>= 1){
      if (tid < off){
        if (rv[tid + off] > rv[tid] ||
            (rv[tid + off] == rv[tid] && ri[tid + off] < ri[tid])){
          rv[tid] = rv[tid + off]; ri[tid] = ri[tid + off];
        }
      }
      __syncthreads();
    }
    far = ri[0];
    __syncthreads();
  }
}

__global__ void k_gather3(const float* __restrict__ xyz, const int* __restrict__ idx,
                          int n, int S, float* __restrict__ out){
  int t = blockIdx.x * blockDim.x + threadIdx.x;
  if (t >= Bc * S) return;
  int s = t % S, b = t / S;
  int j = idx[b * S + s];
  const float* p = xyz + ((size_t)b * n + j) * 3;
  float* o = out + ((size_t)b * S + s) * 3;
  o[0] = p[0]; o[1] = p[1]; o[2] = p[2];
}

// first-k-by-index within radius (PointNet++ convention)
__global__ void k_ball(const float* __restrict__ xyz, const float* __restrict__ ctr,
                       int n, int S, int k, float r2, int* __restrict__ idx){
  int t = blockIdx.x * blockDim.x + threadIdx.x;
  if (t >= Bc * S) return;
  int s = t % S, b = t / S;
  const float* c = ctr + ((size_t)b * S + s) * 3;
  float cx = c[0], cy = c[1], cz = c[2];
  const float* X = xyz + (size_t)b * n * 3;
  int a[32]; int cnt = 0;
  for (int j = 0; j < n && cnt < k; j++){
    float dx = X[j * 3 + 0] - cx, dy = X[j * 3 + 1] - cy, dz = X[j * 3 + 2] - cz;
    float dd = dx * dx + dy * dy + dz * dz;
    if (dd <= r2) a[cnt++] = j;
  }
  int fill = cnt ? a[0] : (n - 1);
  for (int i = cnt; i < k; i++) a[i] = fill;
  int* o = idx + ((size_t)b * S + s) * k;
  for (int i = 0; i < k; i++) o[i] = a[i];
}

// k nearest neighbors by insertion selection (k <= 64)
__global__ void k_knn(const float* __restrict__ ref, const float* __restrict__ q,
                      int n, int S, int k, int* __restrict__ idx){
  int t = blockIdx.x * blockDim.x + threadIdx.x;
  if (t >= Bc * S) return;
  int s = t % S, b = t / S;
  const float* qp = q + ((size_t)b * S + s) * 3;
  float qx = qp[0], qy = qp[1], qz = qp[2];
  const float* X = ref + (size_t)b * n * 3;
  float bd[64]; int bi[64];
  for (int i = 0; i < k; i++){ bd[i] = 3.4e38f; bi[i] = 0; }
  for (int j = 0; j < n; j++){
    float dx = X[j * 3 + 0] - qx, dy = X[j * 3 + 1] - qy, dz = X[j * 3 + 2] - qz;
    float dd = dx * dx + dy * dy + dz * dz;
    if (dd < bd[k - 1]){
      int p = k - 1;
      while (p > 0 && bd[p - 1] > dd){ bd[p] = bd[p - 1]; bi[p] = bi[p - 1]; p--; }
      bd[p] = dd; bi[p] = j;
    }
  }
  int* o = idx + ((size_t)b * S + s) * k;
  for (int i = 0; i < k; i++) o[i] = bi[i];
}

// A row = [nbr_xyz - center (3), feat_nbr (C), zero pad to Kpad]  -> f16
__global__ void k_build_saA(const float* __restrict__ xyz, const float* __restrict__ feat,
                            const float* __restrict__ ctr, const int* __restrict__ idx,
                            int n, int S, int k, int C, int Kpad,
                            half_t* __restrict__ A, int M){
  int r = blockIdx.x * blockDim.x + threadIdx.x;
  if (r >= M) return;
  int ki = r % k; int s = (r / k) % S; int b = r / (k * S);
  int j = idx[((size_t)b * S + s) * k + ki];
  const float* P  = xyz + ((size_t)b * n + j) * 3;
  const float* Cn = ctr + ((size_t)b * S + s) * 3;
  half_t* row = A + (size_t)r * Kpad;
  row[0] = (half_t)(P[0] - Cn[0]);
  row[1] = (half_t)(P[1] - Cn[1]);
  row[2] = (half_t)(P[2] - Cn[2]);
  const float* F = feat + ((size_t)b * n + j) * C;
  for (int c = 0; c < C; c++) row[3 + c] = (half_t)F[c];
  for (int c = 3 + C; c < Kpad; c++) row[c] = (half_t)0.f;
}

// flow-embed A row = [feat1 (C1), feat2_nbr (C2), xyz2_nbr - q (3), pad]
__global__ void k_build_feA(const float* __restrict__ xyz2, const float* __restrict__ q,
                            const float* __restrict__ f1, int C1,
                            const float* __restrict__ f2, int C2,
                            const int* __restrict__ idx, int n, int S, int k, int Kpad,
                            half_t* __restrict__ A, int M){
  int r = blockIdx.x * blockDim.x + threadIdx.x;
  if (r >= M) return;
  int ki = r % k; int s = (r / k) % S; int b = r / (k * S);
  int j = idx[((size_t)b * S + s) * k + ki];
  half_t* row = A + (size_t)r * Kpad;
  const float* a = f1 + ((size_t)b * S + s) * C1;
  for (int c = 0; c < C1; c++) row[c] = (half_t)a[c];
  const float* bb = f2 + ((size_t)b * n + j) * C2;
  for (int c = 0; c < C2; c++) row[C1 + c] = (half_t)bb[c];
  const float* P  = xyz2 + ((size_t)b * n + j) * 3;
  const float* Q  = q + ((size_t)b * S + s) * 3;
  row[C1 + C2 + 0] = (half_t)(P[0] - Q[0]);
  row[C1 + C2 + 1] = (half_t)(P[1] - Q[1]);
  row[C1 + C2 + 2] = (half_t)(P[2] - Q[2]);
  for (int c = C1 + C2 + 3; c < Kpad; c++) row[c] = (half_t)0.f;
}

// A row = [fa (C1), fb (C2), pad] for mlp3 stages
__global__ void k_build_cat2A(const float* __restrict__ fa, int C1,
                              const float* __restrict__ fb, int C2,
                              int S, int Kpad, half_t* __restrict__ A, int M){
  int r = blockIdx.x * blockDim.x + threadIdx.x;
  if (r >= M) return;
  int s = r % S, b = r / S;
  half_t* row = A + (size_t)r * Kpad;
  const float* a = fa + ((size_t)b * S + s) * C1;
  for (int c = 0; c < C1; c++) row[c] = (half_t)a[c];
  const float* bb = fb + ((size_t)b * S + s) * C2;
  for (int c = 0; c < C2; c++) row[C1 + c] = (half_t)bb[c];
  for (int c = C1 + C2; c < Kpad; c++) row[c] = (half_t)0.f;
}

__global__ void k_concat2(const float* __restrict__ a, int C1,
                          const float* __restrict__ b_, int C2,
                          int S, float* __restrict__ out){
  int C = C1 + C2;
  int t = blockIdx.x * blockDim.x + threadIdx.x;
  if (t >= Bc * S * C) return;
  int c = t % C; int s = (t / C) % S; int b = t / (C * S);
  float v = (c < C1) ? a[((size_t)b * S + s) * C1 + c]
                     : b_[((size_t)b * S + s) * C2 + (c - C1)];
  out[((size_t)b * S + s) * C + c] = v;
}

// weight [K,N] f32 -> transposed padded [Nstore][Kpad] f16
__global__ void k_wconv(const float* __restrict__ W, half_t* __restrict__ Wt,
                        int K, int N, int Kpad, int Nst){
  int t = blockIdx.x * blockDim.x + threadIdx.x;
  if (t >= Nst * Kpad) return;
  int n = t / Kpad, k = t % Kpad;
  float v = (n < N && k < K) ? W[(size_t)k * N + n] : 0.f;
  Wt[t] = (half_t)v;
}

union H16 { v16h v; v4f f[2]; };

// D = act(A x W + bias); each wave owns a 16x(16*NB) tile so every A fragment
// is reused by NB v_wmma_f32_16x16x32_f16 issues (register N-blocking).
template<int NB>
__global__ void k_gemm_t(const half_t* __restrict__ A, int lda,
                         const half_t* __restrict__ Wt, int ldw,
                         half_t* __restrict__ O, int ldo,
                         const float* __restrict__ bias, int nbias,
                         int M, int Nst, int Kpad, int relu){
  int lane = threadIdx.x & 31;
  int wave = (blockIdx.x * blockDim.x + threadIdx.x) >> 5;
  int nwav = (gridDim.x * blockDim.x) >> 5;
  int hs = lane >> 4, r = lane & 15;
  int nt = Nst / (16 * NB);
  int tiles = (M >> 4) * nt;
  for (int t = wave; t < tiles; t += nwav){
    int m0 = (t / nt) << 4;
    int n0 = (t % nt) * (16 * NB);
    v8f acc[NB];
    #pragma unroll
    for (int j = 0; j < NB; j++) acc[j] = {};
    const half_t* ap  = A  + (size_t)(m0 + r) * lda + hs * 8;
    const half_t* bp0 = Wt + (size_t)(n0 + r) * ldw + hs * 8;
    for (int kk = 0; kk < Kpad; kk += 32){
      H16 a;
      a.f[0] = *(const v4f*)(ap);
      a.f[1] = *(const v4f*)(ap + 16);
      ap += 32;
      #pragma unroll
      for (int j = 0; j < NB; j++){
        H16 bm;
        const half_t* bp = bp0 + (size_t)j * 16 * ldw;
        bm.f[0] = *(const v4f*)(bp);
        bm.f[1] = *(const v4f*)(bp + 16);
        acc[j] = __builtin_amdgcn_wmma_f32_16x16x32_f16(false, a.v, false, bm.v,
                                                        (short)0, acc[j], false, false);
      }
      bp0 += 32;
    }
    #pragma unroll
    for (int j = 0; j < NB; j++){
      int col = n0 + j * 16 + r;
      float bv = 0.f;
      if (bias && col < nbias) bv = bias[col];
      #pragma unroll
      for (int i = 0; i < 8; i++){
        float v = acc[j][i] + bv;
        if (relu) v = fmaxf(v, 0.f);
        O[(size_t)(m0 + hs * 8 + i) * ldo + col] = (half_t)v;
      }
    }
  }
}

// max over k neighbors: in f16 [B*S*k][ld] -> out f32 [B,S,C]
__global__ void k_maxpool(const half_t* __restrict__ Ain, int ld,
                          int S, int k, int C, float* __restrict__ out){
  int t = blockIdx.x * blockDim.x + threadIdx.x;
  if (t >= Bc * S * C) return;
  int c = t % C; int s = (t / C) % S; int b = t / (C * S);
  const half_t* base = Ain + (size_t)(((size_t)b * S + s) * k) * ld + c;
  float m = -3.4e38f;
  for (int ki = 0; ki < k; ki++) m = fmaxf(m, (float)base[(size_t)ki * ld]);
  out[((size_t)b * S + s) * C + c] = m;
}

// softmax over ball neighbors -> relocated centers
__global__ void k_attn_center(const half_t* __restrict__ Sc, int lds_,
                              const float* __restrict__ xyz, const int* __restrict__ idx,
                              int n, int S, int k, float* __restrict__ out){
  int t = blockIdx.x * blockDim.x + threadIdx.x;
  if (t >= Bc * S) return;
  int s = t % S, b = t / S;
  size_t base = ((size_t)b * S + s) * k;
  float w[32];
  float mx = -3.4e38f;
  for (int ki = 0; ki < k; ki++){
    float v = (float)Sc[(base + ki) * lds_];
    w[ki] = v; if (v > mx) mx = v;
  }
  float sum = 0.f;
  for (int ki = 0; ki < k; ki++){ w[ki] = __expf(w[ki] - mx); sum += w[ki]; }
  float inv = 1.f / sum;
  float ox = 0.f, oy = 0.f, oz = 0.f;
  const float* X = xyz + (size_t)b * n * 3;
  for (int ki = 0; ki < k; ki++){
    int j = idx[base + ki];
    float ww = w[ki] * inv;
    ox += ww * X[j * 3 + 0]; oy += ww * X[j * 3 + 1]; oz += ww * X[j * 3 + 2];
  }
  float* o = out + ((size_t)b * S + s) * 3;
  o[0] = ox; o[1] = oy; o[2] = oz;
}

// 3-NN inverse-distance feature interpolation
__global__ void k_featprop(const float* __restrict__ p, const float* __restrict__ cp,
                           const float* __restrict__ f2, int n, int S2, int C,
                           float* __restrict__ out){
  int t = blockIdx.x * blockDim.x + threadIdx.x;
  if (t >= Bc * n) return;
  int i = t % n, b = t / n;
  const float* q = p + ((size_t)b * n + i) * 3;
  float qx = q[0], qy = q[1], qz = q[2];
  const float* X = cp + (size_t)b * S2 * 3;
  float bd[3] = {3.4e38f, 3.4e38f, 3.4e38f};
  int   bi[3] = {0, 0, 0};
  for (int j = 0; j < S2; j++){
    float dx = X[j * 3 + 0] - qx, dy = X[j * 3 + 1] - qy, dz = X[j * 3 + 2] - qz;
    float dd = fmaxf(dx * dx + dy * dy + dz * dz, 0.f);
    if (dd < bd[2]){
      if (dd < bd[0]){ bd[2]=bd[1]; bi[2]=bi[1]; bd[1]=bd[0]; bi[1]=bi[0]; bd[0]=dd; bi[0]=j; }
      else if (dd < bd[1]){ bd[2]=bd[1]; bi[2]=bi[1]; bd[1]=dd; bi[1]=j; }
      else { bd[2]=dd; bi[2]=j; }
    }
  }
  float w0 = 1.f / (bd[0] + 1e-10f), w1 = 1.f / (bd[1] + 1e-10f), w2 = 1.f / (bd[2] + 1e-10f);
  float inv = 1.f / (w0 + w1 + w2);
  w0 *= inv; w1 *= inv; w2 *= inv;
  const float* F = f2 + (size_t)b * S2 * C;
  float* o = out + ((size_t)b * n + i) * C;
  for (int c = 0; c < C; c++)
    o[c] = w0 * F[(size_t)bi[0] * C + c] + w1 * F[(size_t)bi[1] * C + c] + w2 * F[(size_t)bi[2] * C + c];
}

__global__ void k_meanflow(const float* __restrict__ flow, const int* __restrict__ idx,
                           int n, int S, int k, float* __restrict__ out){
  int t = blockIdx.x * blockDim.x + threadIdx.x;
  if (t >= Bc * S * 3) return;
  int c = t % 3; int s = (t / 3) % S; int b = t / (3 * S);
  size_t base = ((size_t)b * S + s) * k;
  float sum = 0.f;
  for (int ki = 0; ki < k; ki++)
    sum += flow[((size_t)b * n + idx[base + ki]) * 3 + c];
  out[((size_t)b * S + s) * 3 + c] = sum / (float)k;
}

__global__ void k_add(const float* __restrict__ a, const float* __restrict__ b,
                      float* __restrict__ o, int total){
  int t = blockIdx.x * blockDim.x + threadIdx.x;
  if (t < total) o[t] = a[t] + b[t];
}

// f16 [rows][ldo] columns [0,C) -> f32 [rows][C]
__global__ void k_copy_cols(const half_t* __restrict__ O, int ldo, int C,
                            float* __restrict__ dst, int rows){
  int t = blockIdx.x * blockDim.x + threadIdx.x;
  if (t >= rows * C) return;
  dst[t] = (float)O[(size_t)(t / C) * ldo + (t % C)];
}

__global__ void k_copy_mask(const half_t* __restrict__ O, int ldo,
                            float* __restrict__ dst, int rows){
  int t = blockIdx.x * blockDim.x + threadIdx.x;
  if (t < rows) dst[t] = (float)O[(size_t)t * ldo];
}

// ======================= host =======================

struct WtD { half_t* dev; int K, N, Kpad, Nst; };

static void gemm_l(hipStream_t st, const half_t* A, int lda, const WtD& w,
                   half_t* O, int M, const float* bias, int nbias, int relu){
  int NB = (w.Nst % 64 == 0) ? 4 : ((w.Nst % 32 == 0) ? 2 : 1);
  int tiles = (M / 16) * (w.Nst / (16 * NB));
  int blocks = cdivh(tiles, TPB / 32);
  if (blocks > 4096) blocks = 4096;
  if (blocks < 1) blocks = 1;
  if (NB == 4)
    k_gemm_t<4><<<blocks, TPB, 0, st>>>(A, lda, w.dev, w.Kpad, O, w.Nst,
                                        bias, nbias, M, w.Nst, w.Kpad, relu);
  else if (NB == 2)
    k_gemm_t<2><<<blocks, TPB, 0, st>>>(A, lda, w.dev, w.Kpad, O, w.Nst,
                                        bias, nbias, M, w.Nst, w.Kpad, relu);
  else
    k_gemm_t<1><<<blocks, TPB, 0, st>>>(A, lda, w.dev, w.Kpad, O, w.Nst,
                                        bias, nbias, M, w.Nst, w.Kpad, relu);
}

static const half_t* mlp_l(hipStream_t st, const half_t* A0, int lda, int M,
                           const WtD* ws, int nw, half_t* O1, half_t* O2, int* ld_out){
  const half_t* cur = A0; int ld = lda;
  half_t* bufs[2] = {O1, O2};
  for (int i = 0; i < nw; i++){
    gemm_l(st, cur, ld, ws[i], bufs[i & 1], M, nullptr, 0, 1);
    cur = bufs[i & 1]; ld = ws[i].Nst;
  }
  *ld_out = ld;
  return cur;
}

enum { W_SA1 = 0, W_ATTN0 = 3, W_ATTN1 = 4, W_SA2 = 5, W_SA3 = 8, W_SA4 = 11,
       W_FE = 14, W_SU1M2 = 17, W_SU2 = 19, W_SU2M2 = 22, W_SU3 = 23,
       W_SU3M2 = 26, W_FP = 27, W_W1 = 29, W_W2 = 30, W_W3 = 31, W_W4 = 32 };

#define GL(kern, total, ...) kern<<<cdivh((total), TPB), TPB, 0, st>>>(__VA_ARGS__)

extern "C" void kernel_launch(void* const* d_in, const int* in_sizes, int n_in,
                              void* d_out, int out_size, void* d_ws, size_t ws_size,
                              hipStream_t stream) {
  (void)in_sizes; (void)n_in; (void)out_size; (void)ws_size;
  hipStream_t st = stream;
  float* out = (float*)d_out;

  // ---- bump allocator over workspace ----
  char* wsp = (char*)d_ws;
  size_t woff = 0;
  auto alloc = [&](size_t bytes) -> void* {
    void* p = wsp + woff;
    woff += (bytes + 255) & ~(size_t)255;
    return p;
  };

  // ---- convert all weight matrices to transposed/padded f16 ----
  static const int WSPEC[33][3] = {
    {4,6,32},{5,32,32},{6,32,64},{7,67,64},{8,64,1},{9,67,64},{10,64,64},{11,64,128},
    {12,131,128},{13,128,128},{14,128,256},{15,259,256},{16,256,256},{17,256,512},
    {18,259,128},{19,128,128},{20,128,128},{21,771,256},{22,256,256},{23,259,128},
    {24,128,128},{25,128,256},{26,512,256},{27,259,128},{28,128,128},{29,128,256},
    {30,320,256},{31,259,256},{32,256,256},{33,256,128},{34,128,3},{36,256,128},{37,128,1}
  };
  WtD wd[33];
  for (int i = 0; i < 33; i++){
    int K = WSPEC[i][1], N = WSPEC[i][2];
    wd[i].K = K; wd[i].N = N;
    wd[i].Kpad = ruh(K, 32); wd[i].Nst = ruh(N, 16);
    wd[i].dev = (half_t*)alloc((size_t)wd[i].Nst * wd[i].Kpad * sizeof(half_t));
    GL(k_wconv, wd[i].Nst * wd[i].Kpad,
       (const float*)d_in[WSPEC[i][0]], wd[i].dev, K, N, wd[i].Kpad, wd[i].Nst);
  }
  const float* b2 = (const float*)d_in[35];
  const float* b4 = (const float*)d_in[38];

  // ---- persistent feature / geometry buffers ----
  float* p1 = (float*)alloc((size_t)Bc*Nc*3*4);  float* p2 = (float*)alloc((size_t)Bc*Nc*3*4);
  float* f1 = (float*)alloc((size_t)Bc*Nc*3*4);  float* f2 = (float*)alloc((size_t)Bc*Nc*3*4);
  int* fidxbuf = (int*)alloc((size_t)Bc*1024*4);
  int* idxbuf  = (int*)alloc((size_t)Bc*1024*32*4);
  float* l1p1 = (float*)alloc((size_t)Bc*1024*3*4); float* l1p2 = (float*)alloc((size_t)Bc*1024*3*4);
  float* l1f1 = (float*)alloc((size_t)Bc*1024*64*4); float* l1f2 = (float*)alloc((size_t)Bc*1024*64*4);
  float* l2p1f = (float*)alloc((size_t)Bc*256*3*4); float* l2p1 = (float*)alloc((size_t)Bc*256*3*4);
  float* l2p2f = (float*)alloc((size_t)Bc*256*3*4); float* l2p2 = (float*)alloc((size_t)Bc*256*3*4);
  float* l2f1 = (float*)alloc((size_t)Bc*256*128*4); float* l2f2 = (float*)alloc((size_t)Bc*256*128*4);
  float* l2f1n = (float*)alloc((size_t)Bc*256*128*4);
  float* l3p1 = (float*)alloc((size_t)Bc*64*3*4); float* l4p1 = (float*)alloc((size_t)Bc*16*3*4);
  float* l3f1 = (float*)alloc((size_t)Bc*64*256*4); float* l4f1 = (float*)alloc((size_t)Bc*16*512*4);
  float* fsu1 = (float*)alloc((size_t)Bc*64*515*4); float* l3fn = (float*)alloc((size_t)Bc*64*256*4);
  float* fsu2 = (float*)alloc((size_t)Bc*256*256*4);
  float* l2cat = (float*)alloc((size_t)Bc*256*256*4);
  float* l2fn = (float*)alloc((size_t)Bc*256*256*4);
  float* fsu3 = (float*)alloc((size_t)Bc*1024*256*4);
  float* l1fn = (float*)alloc((size_t)Bc*1024*256*4);
  float* interp = (float*)alloc((size_t)Bc*Nc*256*4);
  float* sf1buf = (float*)alloc((size_t)Bc*Nc*3*4);
  float* sf2buf = (float*)alloc((size_t)Bc*Nc*3*4);
  float* cflow = (float*)alloc((size_t)Bc*256*3*4);
  float* qre = (float*)alloc((size_t)Bc*256*3*4);
  half_t* bigA = (half_t*)alloc((size_t)131072 * 288 * sizeof(half_t));
  half_t* O1 = (half_t*)alloc((size_t)16777216 * sizeof(half_t));
  half_t* O2 = (half_t*)alloc((size_t)16777216 * sizeof(half_t));
  half_t* O3 = (half_t*)alloc((size_t)1048576 * sizeof(half_t));

  // ---- generic SA module (FPS -> ball -> group -> MLP4 -> maxpool) ----
  auto do_sa = [&](const float* xyz, const float* feat, int n, int Cf,
                   int np, float r2, int k, int wbase,
                   float* nxyz, float* pooled, int Cout){
    k_fps<<<Bc, TPB, 0, st>>>(xyz, n, np, fidxbuf);
    GL(k_gather3, Bc*np, xyz, fidxbuf, n, np, nxyz);
    GL(k_ball, Bc*np, xyz, nxyz, n, np, k, r2, idxbuf);
    int M = Bc * np * k;
    int Kpad = ruh(3 + Cf, 32);
    GL(k_build_saA, M, xyz, feat, nxyz, idxbuf, n, np, k, Cf, Kpad, bigA, M);
    int ld; const half_t* O = mlp_l(st, bigA, Kpad, M, &wd[wbase], 3, O1, O2, &ld);
    GL(k_maxpool, Bc*np*Cout, O, ld, np, k, Cout, pooled);
  };

  // ---- setupconv (kNN -> group -> [MLP4] -> maxpool -> concat -> MLP3) ----
  auto do_setup = [&](const float* pos1, int S1, const float* pos2, int n2,
                      const float* feat1, int C1, const float* feat2, int C2,
                      int k, int wbase, int nw, int w2base, int nw2,
                      float* poolbuf, float* outF){
    GL(k_knn, Bc*S1, pos2, pos1, n2, S1, k, idxbuf);
    int M = Bc * S1 * k;
    int Kpad = ruh(3 + C2, 32);
    GL(k_build_saA, M, pos2, feat2, pos1, idxbuf, n2, S1, k, C2, Kpad, bigA, M);
    int CP;
    if (nw > 0){
      int ld; const half_t* O = mlp_l(st, bigA, Kpad, M, &wd[wbase], nw, O1, O2, &ld);
      CP = wd[wbase + nw - 1].N;
      GL(k_maxpool, Bc*S1*CP, O, ld, S1, k, CP, poolbuf);
    } else {
      CP = 3 + C2;
      GL(k_maxpool, Bc*S1*CP, bigA, Kpad, S1, k, CP, poolbuf);
    }
    int M2 = Bc * S1;
    int K2p = ruh(CP + C1, 32);
    GL(k_build_cat2A, M2, poolbuf, CP, feat1, C1, S1, K2p, bigA, M2);
    int ld2; const half_t* Of = mlp_l(st, bigA, K2p, M2, &wd[w2base], nw2, O1, O2, &ld2);
    int Co = wd[w2base + nw2 - 1].N;
    GL(k_copy_cols, M2*Co, Of, ld2, Co, outF, M2);
  };

  // ---- SA1 + attention SA2 (shared by pc1 / pc2) ----
  auto do_l12 = [&](const float* p, const float* f, float* l1p, float* l1f,
                    float* l2pf, float* l2p, float* l2f){
    do_sa(p, f, Nc, 3, 1024, 1.0f, 32, W_SA1, l1p, l1f, 64);
    // sa_attn: FPS candidates, softmax-relocated centers, regroup + MLP
    k_fps<<<Bc, TPB, 0, st>>>(l1p, 1024, 256, fidxbuf);
    GL(k_gather3, Bc*256, l1p, fidxbuf, 1024, 256, l2pf);
    GL(k_ball, Bc*256, l1p, l2pf, 1024, 256, 32, 4.0f, idxbuf);
    int M = Bc * 256 * 32;
    GL(k_build_saA, M, l1p, l1f, l2pf, idxbuf, 1024, 256, 32, 64, 96, bigA, M);
    gemm_l(st, bigA, 96, wd[W_ATTN0], O1, M, nullptr, 0, 1);
    gemm_l(st, O1, wd[W_ATTN0].Nst, wd[W_ATTN1], O2, M, nullptr, 0, 0);
    GL(k_attn_center, Bc*256, O2, wd[W_ATTN1].Nst, l1p, idxbuf, 1024, 256, 32, l2p);
    GL(k_ball, Bc*256, l1p, l2p, 1024, 256, 32, 4.0f, idxbuf);
    GL(k_build_saA, M, l1p, l1f, l2p, idxbuf, 1024, 256, 32, 64, 96, bigA, M);
    int ld; const half_t* O = mlp_l(st, bigA, 96, M, &wd[W_SA2], 3, O1, O2, &ld);
    GL(k_maxpool, Bc*256*128, O, ld, 256, 32, 128, l2f);
  };

  // ---- refinement pass ----
  auto refine = [&](const float* qxyz, float* sfbuf, float* maskdst){
    // flow embedding: kNN(64) of queries in pc2's level-2 cloud
    GL(k_knn, Bc*256, l2p2, qxyz, 256, 256, 64, idxbuf);
    int M = Bc * 256 * 64;
    GL(k_build_feA, M, l2p2, qxyz, l2f1, 128, l2f2, 128, idxbuf, 256, 256, 64, 288, bigA, M);
    int ld; const half_t* O = mlp_l(st, bigA, 288, M, &wd[W_FE], 3, O1, O2, &ld);
    GL(k_maxpool, Bc*256*128, O, ld, 256, 64, 128, l2f1n);
    // downstream SA pyramid
    do_sa(l2p1, l2f1n, 256, 128, 64, 4.0f, 8, W_SA3, l3p1, l3f1, 256);
    do_sa(l3p1, l3f1, 64, 256, 16, 16.0f, 8, W_SA4, l4p1, l4f1, 512);
    // upconv chain
    do_setup(l3p1, 64, l4p1, 16, l3f1, 256, l4f1, 512, 8, 0, 0, W_SU1M2, 2, fsu1, l3fn);
    GL(k_concat2, Bc*256*256, l2f1, 128, l2f1n, 128, 256, l2cat);
    do_setup(l2p1, 256, l3p1, 64, l2cat, 256, l3fn, 256, 8, W_SU2, 3, W_SU2M2, 1, fsu2, l2fn);
    do_setup(l1p1, 1024, l2p1, 256, l1f1, 64, l2fn, 256, 8, W_SU3, 3, W_SU3M2, 1, fsu3, l1fn);
    // feature propagation to full resolution
    GL(k_featprop, Bc*Nc, p1, l1p1, l1fn, Nc, 1024, 256, interp);
    int M2 = Bc * Nc;
    GL(k_build_cat2A, M2, interp, 256, f1, 3, Nc, 288, bigA, M2);
    int ldf; const half_t* L0 = mlp_l(st, bigA, 288, M2, &wd[W_FP], 2, O1, O2, &ldf);
    // heads: scene flow + mask
    gemm_l(st, L0, ldf, wd[W_W1], O1, M2, nullptr, 0, 1);
    gemm_l(st, O1, wd[W_W1].Nst, wd[W_W2], O3, M2, b2, 3, 0);
    GL(k_copy_cols, M2*3, O3, wd[W_W2].Nst, 3, sfbuf, M2);
    gemm_l(st, L0, ldf, wd[W_W3], O1, M2, nullptr, 0, 1);
    gemm_l(st, O1, wd[W_W3].Nst, wd[W_W4], O3, M2, b4, 1, 0);
    GL(k_copy_mask, M2, O3, wd[W_W4].Nst, maskdst, M2);
  };

  // ---- output offsets (reference tuple order, flattened) ----
  size_t off = 0;
  size_t off_sf1  = off; off += (size_t)Bc*3*Nc;
  size_t off_sf2  = off; off += (size_t)Bc*3*Nc;
  size_t off_m1   = off; off += (size_t)Bc*Nc;
  size_t off_m2   = off; off += (size_t)Bc*Nc;
  size_t off_l1p1f= off; off += (size_t)Bc*3*1024;
  size_t off_l1p1 = off; off += (size_t)Bc*3*1024;
  size_t off_l2p1f= off; off += (size_t)Bc*3*256;
  size_t off_l2p1 = off; off += (size_t)Bc*3*256;
  size_t off_l1p2f= off; off += (size_t)Bc*3*1024;
  size_t off_l1p2 = off; off += (size_t)Bc*3*1024;
  size_t off_l2p2f= off; off += (size_t)Bc*3*256;
  size_t off_l2p2 = off; off += (size_t)Bc*3*256;
  size_t off_l3p1 = off; off += (size_t)Bc*3*64;
  size_t off_l4p1 = off; off += (size_t)Bc*3*16;

  // ---- pipeline ----
  GL(k_tin, Bc*3*Nc, (const float*)d_in[0], p1, Nc);
  GL(k_tin, Bc*3*Nc, (const float*)d_in[1], p2, Nc);
  GL(k_tin, Bc*3*Nc, (const float*)d_in[2], f1, Nc);
  GL(k_tin, Bc*3*Nc, (const float*)d_in[3], f2, Nc);

  do_l12(p1, f1, l1p1, l1f1, l2p1f, l2p1, l2f1);
  do_l12(p2, f2, l1p2, l1f2, l2p2f, l2p2, l2f2);

  refine(l2p1, sf1buf, out + off_m1);
  GL(k_tout3, Bc*3*Nc,  sf1buf, out + off_sf1, Nc);
  GL(k_tout3, Bc*3*64,  l3p1,   out + off_l3p1, 64);
  GL(k_tout3, Bc*3*16,  l4p1,   out + off_l4p1, 16);

  // coarse flow fusion -> relocated queries -> second refinement
  GL(k_ball, Bc*256, p1, l2p1, Nc, 256, 32, 4.0f, idxbuf);
  GL(k_meanflow, Bc*256*3, sf1buf, idxbuf, Nc, 256, 32, cflow);
  GL(k_add, Bc*256*3, cflow, l2p1, qre, Bc*256*3);
  refine(qre, sf2buf, out + off_m2);
  GL(k_tout3, Bc*3*Nc, sf2buf, out + off_sf2, Nc);

  // remaining geometry outputs
  GL(k_tout3, Bc*3*1024, l1p1,  out + off_l1p1f, 1024);
  GL(k_tout3, Bc*3*1024, l1p1,  out + off_l1p1,  1024);
  GL(k_tout3, Bc*3*256,  l2p1f, out + off_l2p1f, 256);
  GL(k_tout3, Bc*3*256,  l2p1,  out + off_l2p1,  256);
  GL(k_tout3, Bc*3*1024, l1p2,  out + off_l1p2f, 1024);
  GL(k_tout3, Bc*3*1024, l1p2,  out + off_l1p2,  1024);
  GL(k_tout3, Bc*3*256,  l2p2f, out + off_l2p2f, 256);
  GL(k_tout3, Bc*3*256,  l2p2,  out + off_l2p2,  256);
}
